// MultiHeadAttention_11605001633995
// MI455X (gfx1250) — matrix operand order repaired and checked by reading, not compiled, not run on hardware
//
#include <hip/hip_runtime.h>
#include <hip/hip_bf16.h>

// ---------------------------------------------------------------- types
typedef __attribute__((ext_vector_type(16))) __bf16 bf16x16;
typedef __attribute__((ext_vector_type(8)))  __bf16 bf16x8;
typedef __attribute__((ext_vector_type(8)))  float  f32x8;

#define DEV static __device__ __forceinline__

// Problem constants (match reference)
#define B_  2
#define T_  2048
#define H_  16
#define DH_ 64
#define DIN_  1024
#define DOUT_ 1024
#define M_  (B_ * T_)          // 4096 rows of x / y

// ---------------------------------------------------------------- WMMA helpers
// A fragment (16x32 bf16, MxK), source row-major [rows][ld]:
//  lane<16 : M=lane,    elems 0..15 -> K=k0..k0+15
//  lane>=16: M=lane-16, elems 0..15 -> K=k0+16..k0+31
DEV bf16x16 load_frag_a(const __bf16* __restrict__ p, int row0, int ld, int k0) {
  const int lane = threadIdx.x & 31;
  const int half = lane >> 4;
  const int r    = lane & 15;
  const __bf16* base = p + (size_t)(row0 + r) * (size_t)ld + (k0 + half * 16);
  bf16x8 lo = *(const bf16x8*)(base);
  bf16x8 hi = *(const bf16x8*)(base + 8);
  bf16x16 f;
#pragma unroll
  for (int i = 0; i < 8; ++i) { f[i] = lo[i]; f[i + 8] = hi[i]; }
  return f;
}

// B fragment (32x16 bf16, KxN), source is "B transposed" row-major [n][ld over k]:
//  lane<16 : N=lane,    elems 0..15 -> K=k0..k0+15
//  lane>=16: N=lane-16, elems 0..15 -> K=k0+16..k0+31
DEV bf16x16 load_frag_b(const __bf16* __restrict__ p, int n0, int ld, int k0) {
  const int lane = threadIdx.x & 31;
  const int half = lane >> 4;
  const int n    = lane & 15;
  const __bf16* base = p + (size_t)(n0 + n) * (size_t)ld + (k0 + half * 16);
  bf16x8 lo = *(const bf16x8*)(base);
  bf16x8 hi = *(const bf16x8*)(base + 8);
  bf16x16 f;
#pragma unroll
  for (int i = 0; i < 8; ++i) { f[i] = lo[i]; f[i + 8] = hi[i]; }
  return f;
}

DEV f32x8 wmma_bf16(bf16x16 a, bf16x16 b, f32x8 c) {
  return __builtin_amdgcn_wmma_f32_16x16x32_bf16(false, a, false, b, (short)0, c,
                                                 false, false);
}

// ---------------------------------------------------------------- converts
__global__ void cvt_f32_to_bf16(const float* __restrict__ in,
                                __bf16* __restrict__ out, int n) {
  int i = blockIdx.x * blockDim.x + threadIdx.x;
  if (i < n) out[i] = (__bf16)in[i];
}

// W [K][N] f32  ->  Wt [N][K] bf16
__global__ void cvt_transpose_bf16(const float* __restrict__ W,
                                   __bf16* __restrict__ Wt, int K, int N) {
  int i = blockIdx.x * blockDim.x + threadIdx.x;
  if (i >= K * N) return;
  int k = i / N, n = i % N;
  Wt[(size_t)n * K + k] = (__bf16)W[(size_t)k * N + n];
}

// ---------------------------------------------------------------- GEMM
// C[M][N] = A[M][K] * Bt[N][K]^T ; per-wave 32x64 tile (2x4 WMMA tiles).
// ~130 VGPRs/lane (no spills), 24 B/lane of fragment traffic per WMMA.
// __launch_bounds__(256,1): matrix-bound kernel, allow full VGPR budget.
// mode 0: store bf16 Q/K layout [B*H][T][64]
// mode 1: store bf16 V^T layout [B*H][64][T]
// mode 2: store f32 [M][N] with +bias[n]  (final projection)
__global__ void __launch_bounds__(256, 1)
gemm_wmma(const __bf16* __restrict__ A,
          const __bf16* __restrict__ Bt,
          const float* __restrict__ bias,
          void* __restrict__ outp,
          int M, int N, int Kd, int mode) {
  const int w = (int)((blockIdx.x * blockDim.x + threadIdx.x) >> 5);
  const int tilesN = N >> 6;
  const int mt = w / tilesN, nt = w % tilesN;
  const int m0 = mt * 32, n0 = nt * 64;
  if (m0 >= M) return;  // wave-uniform

  f32x8 acc[2][4];
#pragma unroll
  for (int i = 0; i < 2; ++i)
#pragma unroll
    for (int j = 0; j < 4; ++j) acc[i][j] = (f32x8){};

  for (int k0 = 0; k0 < Kd; k0 += 32) {
    bf16x16 a[2], b[4];
#pragma unroll
    for (int i = 0; i < 2; ++i) a[i] = load_frag_a(A, m0 + 16 * i, Kd, k0);
#pragma unroll
    for (int j = 0; j < 4; ++j) b[j] = load_frag_b(Bt, n0 + 16 * j, Kd, k0);
#pragma unroll
    for (int i = 0; i < 2; ++i)
#pragma unroll
      for (int j = 0; j < 4; ++j)
        acc[i][j] = wmma_bf16(a[i], b[j], acc[i][j]);
  }

  const int lane = threadIdx.x & 31;
  const int half = lane >> 4, nl = lane & 15;
  __bf16* ob = (__bf16*)outp;
  float*  of = (float*)outp;

  auto emit = [&](const f32x8& c, int mb, int nb) {
#pragma unroll
    for (int r = 0; r < 8; ++r) {
      int m = mb + r + 8 * half;
      int n = nb + nl;
      float v = c[r];
      if (mode == 0) {                       // [B*H][T][64]
        int bb = m >> 11, t = m & (T_ - 1), h = n >> 6, d = n & 63;
        ob[(((size_t)(bb * H_ + h) * T_) + t) * DH_ + d] = (__bf16)v;
      } else if (mode == 1) {                // [B*H][64][T]
        int bb = m >> 11, t = m & (T_ - 1), h = n >> 6, d = n & 63;
        ob[(((size_t)(bb * H_ + h) * DH_) + d) * T_ + t] = (__bf16)v;
      } else {                               // f32 + bias
        of[(size_t)m * N + n] = v + bias[n];
      }
    }
  };
#pragma unroll
  for (int i = 0; i < 2; ++i)
#pragma unroll
    for (int j = 0; j < 4; ++j)
      emit(acc[i][j], m0 + 16 * i, n0 + 16 * j);
}

// ---------------------------------------------------------------- attention
// One wave per (b, h, 16-query tile). Computes S^T = K*Q^T (so each lane owns
// one query column -> cheap softmax reductions), online softmax over 32-key
// blocks, O^T = V^T * P^T via WMMA.
__global__ void attn_wmma(const __bf16* __restrict__ Q,
                          const __bf16* __restrict__ K,
                          const __bf16* __restrict__ Vt,
                          __bf16* __restrict__ Y) {
  const int w  = (int)((blockIdx.x * blockDim.x + threadIdx.x) >> 5);
  const int qt = w & (T_ / 16 - 1);   // query tile 0..127
  const int bh = w >> 7;              // 0..31
  if (bh >= B_ * H_) return;          // wave-uniform

  const int lane = threadIdx.x & 31;
  const int half = lane >> 4;
  const int il   = lane & 15;         // this lane's query index within tile

  const __bf16* Qp = Q  + (size_t)bh * T_ * DH_;
  const __bf16* Kp = K  + (size_t)bh * T_ * DH_;
  const __bf16* Vp = Vt + (size_t)bh * DH_ * T_;

  // Q^T as B fragments (n = query, k = d), two k-steps cover Dh=64
  bf16x16 bq0 = load_frag_b(Qp, qt * 16, DH_, 0);
  bf16x16 bq1 = load_frag_b(Qp, qt * 16, DH_, 32);

  f32x8 o0 = {}, o1 = {}, o2 = {}, o3 = {};   // O^T, d = 0..63
  float row_max = -3.0e38f;
  float row_sum = 0.f;
  const float scale = 0.125f;                 // 1/sqrt(64)
  const int i_glob = qt * 16 + il;

  const int nblocks = (qt >> 1) + 1;          // 32 keys per block, causal bound
  for (int blk = 0; blk < nblocks; ++blk) {
    const int jb = blk * 32;

    // S^T tiles: rows = key (local), cols = query
    f32x8 s0 = {}, s1 = {};
    bf16x16 ak;
    ak = load_frag_a(Kp, jb,      DH_, 0);  s0 = wmma_bf16(ak, bq0, s0);
    ak = load_frag_a(Kp, jb,      DH_, 32); s0 = wmma_bf16(ak, bq1, s0);
    ak = load_frag_a(Kp, jb + 16, DH_, 0);  s1 = wmma_bf16(ak, bq0, s1);
    ak = load_frag_a(Kp, jb + 16, DH_, 32); s1 = wmma_bf16(ak, bq1, s1);

    // scale + causal mask; per-query (per-lane) block max
    float p0[8], p1[8];
    float mloc = -3.0e38f;
#pragma unroll
    for (int r = 0; r < 8; ++r) {
      int j0 = jb + r + 8 * half;
      int j1 = j0 + 16;
      float v0 = (j0 <= i_glob) ? s0[r] * scale : -3.0e38f;
      float v1 = (j1 <= i_glob) ? s1[r] * scale : -3.0e38f;
      p0[r] = v0; p1[r] = v1;
      mloc = fmaxf(mloc, fmaxf(v0, v1));
    }
    mloc = fmaxf(mloc, __shfl_xor(mloc, 16, 32));
    float newmax = fmaxf(row_max, mloc);
    float alpha  = __expf(row_max - newmax);
    float lsum = 0.f;
#pragma unroll
    for (int r = 0; r < 8; ++r) {
      p0[r] = __expf(p0[r] - newmax);
      p1[r] = __expf(p1[r] - newmax);
      lsum += p0[r] + p1[r];
    }
    lsum += __shfl_xor(lsum, 16, 32);
    row_sum = row_sum * alpha + lsum;
    row_max = newmax;
#pragma unroll
    for (int e = 0; e < 8; ++e) {
      o0[e] *= alpha; o1[e] *= alpha; o2[e] *= alpha; o3[e] *= alpha;
    }

    // Assemble P^T (32 keys x 16 queries) as a B fragment from registers.
    // B wants: lane<16 -> keys jb..jb+15 (tile s0); lane>=16 -> jb+16..jb+31 (s1).
    // Within each tile, key rows r+8 live in the opposite lane-half -> xor-16.
    bf16x16 pb;
#pragma unroll
    for (int e = 0; e < 8; ++e) {
      float q0 = __shfl_xor(p0[e], 16, 32);
      float q1 = __shfl_xor(p1[e], 16, 32);
      bool low = (lane < 16);
      pb[e]     = (__bf16)(low ? p0[e] : q1);
      pb[e + 8] = (__bf16)(low ? q0 : p1[e]);
    }

    // O^T += V^T * P^T  (A rows = d, A k-dim = key; Vt row-major [d][T])
    bf16x16 av;
    av = load_frag_a(Vp,  0, T_, jb); o0 = wmma_bf16(av, pb, o0);
    av = load_frag_a(Vp, 16, T_, jb); o1 = wmma_bf16(av, pb, o1);
    av = load_frag_a(Vp, 32, T_, jb); o2 = wmma_bf16(av, pb, o2);
    av = load_frag_a(Vp, 48, T_, jb); o3 = wmma_bf16(av, pb, o3);
  }

  // Epilogue: divide by row sum (per-lane scalar), write Y[b*T+t][h*64+d] bf16
  float inv = 1.0f / row_sum;
  const int b = bh >> 4, h = bh & 15;
  __bf16* yrow = Y + ((size_t)(b * T_ + qt * 16 + il)) * DOUT_ + h * DH_;
#pragma unroll
  for (int r = 0; r < 8; ++r) {
    int dl = r + 8 * half;
    yrow[dl]      = (__bf16)(o0[r] * inv);
    yrow[16 + dl] = (__bf16)(o1[r] * inv);
    yrow[32 + dl] = (__bf16)(o2[r] * inv);
    yrow[48 + dl] = (__bf16)(o3[r] * inv);
  }
}

// ---------------------------------------------------------------- launch
extern "C" void kernel_launch(void* const* d_in, const int* in_sizes, int n_in,
                              void* d_out, int out_size, void* d_ws, size_t ws_size,
                              hipStream_t stream) {
  (void)in_sizes; (void)n_in; (void)out_size; (void)ws_size;
  const float* x  = (const float*)d_in[0];
  const float* Wq = (const float*)d_in[1];
  const float* Wk = (const float*)d_in[2];
  const float* Wv = (const float*)d_in[3];
  const float* Wo = (const float*)d_in[4];
  const float* bo = (const float*)d_in[5];

  // workspace carve-out (48 MB total, 256B-aligned chunks)
  size_t off = 0;
  auto carve = [&](size_t bytes) -> char* {
    char* p = (char*)d_ws + off;
    off += (bytes + 255) & ~(size_t)255;
    return p;
  };
  __bf16* xb  = (__bf16*)carve((size_t)M_ * DIN_ * 2);      // 8 MB
  __bf16* WqT = (__bf16*)carve((size_t)DIN_ * DOUT_ * 2);   // 2 MB
  __bf16* WkT = (__bf16*)carve((size_t)DIN_ * DOUT_ * 2);
  __bf16* WvT = (__bf16*)carve((size_t)DIN_ * DOUT_ * 2);
  __bf16* WoT = (__bf16*)carve((size_t)DOUT_ * DOUT_ * 2);
  __bf16* Qb  = (__bf16*)carve((size_t)M_ * DOUT_ * 2);     // [B*H][T][64]
  __bf16* Kb  = (__bf16*)carve((size_t)M_ * DOUT_ * 2);
  __bf16* Vtb = (__bf16*)carve((size_t)M_ * DOUT_ * 2);     // [B*H][64][T]
  __bf16* Yb  = (__bf16*)carve((size_t)M_ * DOUT_ * 2);

  // 1) converts
  {
    int n = M_ * DIN_;
    cvt_f32_to_bf16<<<(n + 255) / 256, 256, 0, stream>>>(x, xb, n);
    int nw = DIN_ * DOUT_;
    int g  = (nw + 255) / 256;
    cvt_transpose_bf16<<<g, 256, 0, stream>>>(Wq, WqT, DIN_, DOUT_);
    cvt_transpose_bf16<<<g, 256, 0, stream>>>(Wk, WkT, DIN_, DOUT_);
    cvt_transpose_bf16<<<g, 256, 0, stream>>>(Wv, WvT, DIN_, DOUT_);
    cvt_transpose_bf16<<<g, 256, 0, stream>>>(Wo, WoT, DOUT_, DOUT_);
  }

  // 2) projections: (M/32)*(N/64) waves, 8 waves (256 thr) per block
  {
    int waves  = (M_ / 32) * (DOUT_ / 64);   // 2048
    int blocks = waves * 32 / 256;           // 256
    gemm_wmma<<<blocks, 256, 0, stream>>>(xb, WqT, nullptr, Qb,  M_, DOUT_, DIN_, 0);
    gemm_wmma<<<blocks, 256, 0, stream>>>(xb, WkT, nullptr, Kb,  M_, DOUT_, DIN_, 0);
    gemm_wmma<<<blocks, 256, 0, stream>>>(xb, WvT, nullptr, Vtb, M_, DOUT_, DIN_, 1);
  }

  // 3) attention: B*H*(T/16) = 4096 waves
  {
    int waves  = B_ * H_ * (T_ / 16);
    int blocks = waves * 32 / 256;           // 512
    attn_wmma<<<blocks, 256, 0, stream>>>(Qb, Kb, Vtb, Yb);
  }

  // 4) output projection + bias -> f32 d_out
  {
    int waves  = (M_ / 32) * (DOUT_ / 64);
    int blocks = waves * 32 / 256;
    gemm_wmma<<<blocks, 256, 0, stream>>>(Yb, WoT, bo, d_out, M_, DOUT_, DOUT_, 2);
  }
}
